// TemporalGraphAttention_21492016349927
// MI455X (gfx1250) — compile-verified, matrix operand order checked by reading, not run on hardware
//
#include <hip/hip_runtime.h>
#include <hip/hip_bf16.h>

// ---------------------------------------------------------------------------
// Temporal graph attention, MI455X (gfx1250, wave32, WMMA bf16).
// K/V GEMMs via v_wmma_f32_16x16x32_bf16 with LDS-staged bf16 weights.
// Q computed once per dst node (time-encoding at t=0 is a constant -> folded
// into an effective bias), then gathered per edge from an L2-resident table.
// Streaming inputs use non-temporal loads so the Q table / accumulators stay
// L2-resident; persistent edge blocks amortize the 96KB LDS weight stage and
// prefetch the next tile's rows behind the current tile's WMMAs.
// ---------------------------------------------------------------------------

typedef __attribute__((ext_vector_type(16))) __bf16 v16bf;
typedef __attribute__((ext_vector_type(8)))  float  v8f;
typedef __attribute__((ext_vector_type(4)))  float  v4f;   // clang vector: OK for NT builtins

#define NODE_DIM 128
#define EDGE_DIM 32
#define TIME_DIM 16
#define OUT_DIM  128
#define MSG_DIM  176      // 128 + 32 + 16
#define NKC_MSG  6        // K padded to 192 = 6 * 32
#define NKC_Q    4        // K = 128 = 4 * 32
#define JT       8        // 128 / 16 output col tiles

// B-fragment element mapping mirroring the 16-bit A layout in the ISA:
// lane<16 holds K in {0..7}u{16..23}, lane>=16 holds {8..15}u{24..31}.
__device__ __forceinline__ int frag_kk(int i, int half) {
  return (i & 7) + ((i >> 3) << 4) + (half << 3);
}

__device__ __forceinline__ void load8_nt(const float* p, float v8[8]) {
  const v4f* q = (const v4f*)p;
  v4f x0 = __builtin_nontemporal_load(q);
  v4f x1 = __builtin_nontemporal_load(q + 1);
#pragma unroll
  for (int i = 0; i < 4; ++i) { v8[i] = x0[i]; v8[4 + i] = x1[i]; }
}

// ---------------------------------------------------------------------------
// Kernel 0: repack weights to bf16 fragment-linear layout + effective Q bias.
// ---------------------------------------------------------------------------
__device__ __forceinline__ v16bf pack_frag(const float* W, int ld, int ncols,
                                           int j, int kc, int lane) {
  int half = lane >> 4, n = lane & 15;
  v16bf r;
#pragma unroll
  for (int i = 0; i < 16; ++i) {
    int c = kc * 32 + frag_kk(i, half);
    float v = (c < ncols) ? W[(j * 16 + n) * ld + c] : 0.0f;
    r[i] = (__bf16)v;
  }
  return r;
}

__global__ void prep_kernel(const float* __restrict__ Wq_w,
                            const float* __restrict__ Wq_b,
                            const float* __restrict__ Wk_w,
                            const float* __restrict__ Wv_w,
                            const float* __restrict__ te_phi,
                            v16bf* __restrict__ wkf, v16bf* __restrict__ wvf,
                            v16bf* __restrict__ wqf, float* __restrict__ bq) {
  const int NK = JT * NKC_MSG * 32;   // 1536 per-lane fragments (v16bf each)
  const int NQ = JT * NKC_Q * 32;     // 1024
  int t = blockIdx.x * blockDim.x + threadIdx.x;
  if (t < NK) {
    int j = t / (NKC_MSG * 32), kc = (t / 32) % NKC_MSG, lane = t & 31;
    wkf[t] = pack_frag(Wk_w, MSG_DIM, MSG_DIM, j, kc, lane);
  } else if (t < 2 * NK) {
    int u = t - NK;
    int j = u / (NKC_MSG * 32), kc = (u / 32) % NKC_MSG, lane = u & 31;
    wvf[u] = pack_frag(Wv_w, MSG_DIM, MSG_DIM, j, kc, lane);
  } else if (t < 2 * NK + NQ) {
    int u = t - 2 * NK;
    int j = u / (NKC_Q * 32), kc = (u / 32) % NKC_Q, lane = u & 31;
    wqf[u] = pack_frag(Wq_w, NODE_DIM + TIME_DIM, NODE_DIM, j, kc, lane);
  } else if (t < 2 * NK + NQ + OUT_DIM) {
    int j = t - 2 * NK - NQ;
    // q_te = time_encode(0) = [phi0, sin(phi1..)] : constant -> fold into bias
    float s = Wq_b[j];
#pragma unroll
    for (int tt = 0; tt < TIME_DIM; ++tt) {
      float te = (tt == 0) ? te_phi[0] : __sinf(te_phi[tt]);
      s += Wq_w[j * (NODE_DIM + TIME_DIM) + NODE_DIM + tt] * te;
    }
    bq[j] = s;
  }
}

// ---------------------------------------------------------------------------
// Kernel 1: Q table = dst_feat @ Wq[:, :128]^T + bias_eff   (WMMA bf16)
// ---------------------------------------------------------------------------
__global__ __launch_bounds__(256) void qtable_kernel(
    const float* __restrict__ dst_feat, const v16bf* __restrict__ wqf,
    const float* __restrict__ bq, float* __restrict__ Qt, int n_dst) {
  __shared__ v16bf sW[JT * NKC_Q * 32];   // 32 KB
  __shared__ float sB[OUT_DIM];
  for (int t = threadIdx.x; t < JT * NKC_Q * 32; t += 256) sW[t] = wqf[t];
  for (int t = threadIdx.x; t < OUT_DIM; t += 256) sB[t] = bq[t];
  __syncthreads();

  int lane = threadIdx.x & 31, wv = threadIdx.x >> 5;
  int tile = blockIdx.x * 8 + wv;
  int ntiles = (n_dst + 15) >> 4;
  if (tile >= ntiles) return;
  int half = lane >> 4, n = lane & 15;
  bool full = (tile * 16 + 16 <= n_dst);

  int arow = min(tile * 16 + n, n_dst - 1);      // A: M = lane&15
  v16bf a[NKC_Q];
#pragma unroll
  for (int kc = 0; kc < NKC_Q; ++kc) {
#pragma unroll
    for (int ri = 0; ri < 2; ++ri) {             // runs of 8 contiguous cols
      int base = kc * 32 + ri * 16 + half * 8;
      float v8[8];
      load8_nt(dst_feat + (size_t)arow * NODE_DIM + base, v8);
#pragma unroll
      for (int i = 0; i < 8; ++i) a[kc][ri * 8 + i] = (__bf16)v8[i];
    }
  }
#pragma unroll
  for (int j = 0; j < JT; ++j) {
    v8f acc = {};
#pragma unroll
    for (int kc = 0; kc < NKC_Q; ++kc)
      acc = __builtin_amdgcn_wmma_f32_16x16x32_bf16(
          false, a[kc], false, sW[(j * NKC_Q + kc) * 32 + lane],
          (short)0, acc, false, false);
    float bias = sB[j * 16 + n];
    size_t outb = (size_t)(tile * 16 + 8 * half) * OUT_DIM + j * 16 + n;
    if (full) {
#pragma unroll
      for (int r = 0; r < 8; ++r)                // D: M = r + 8*half, N = n
        Qt[outb + (size_t)r * OUT_DIM] = acc[r] + bias;
    } else {
#pragma unroll
      for (int r = 0; r < 8; ++r) {
        int row = tile * 16 + r + 8 * half;
        if (row < n_dst) Qt[(size_t)row * OUT_DIM + j * 16 + n] = acc[r] + bias;
      }
    }
  }
}

// ---------------------------------------------------------------------------
// Kernel 2: persistent; per-16-edge tile: msg -> (K,V) via WMMA, fused
// scoring, atomic segment sums. Streaming inputs NT; next tile prefetched.
// ---------------------------------------------------------------------------
__global__ __launch_bounds__(256) void edge_kernel(
    const float* __restrict__ src_feat, const float* __restrict__ edge_feat,
    const float* __restrict__ td, const float* __restrict__ te_w,
    const float* __restrict__ te_phi, const v16bf* __restrict__ wkf,
    const v16bf* __restrict__ wvf, const float* __restrict__ Wk_b,
    const float* __restrict__ Wv_b, const int* __restrict__ dst_ids,
    const float* __restrict__ Qt, float* __restrict__ dst_agg,
    float* __restrict__ attn_norm, int E) {
  __shared__ v16bf sWk[JT * NKC_MSG * 32];   // 48 KB
  __shared__ v16bf sWv[JT * NKC_MSG * 32];   // 48 KB
  for (int t = threadIdx.x; t < JT * NKC_MSG * 32; t += 256) {
    sWk[t] = wkf[t];
    sWv[t] = wvf[t];
  }
  __syncthreads();

  int lane = threadIdx.x & 31, wv = threadIdx.x >> 5;
  int half = lane >> 4, n = lane & 15;
  int ntiles = (E + 15) >> 4;
  int stride = gridDim.x * 8;
  const float inv_sqrt_hd = 0.17677669529663687f;   // 1/sqrt(32)

  for (int tile = blockIdx.x * 8 + wv; tile < ntiles; tile += stride) {
    // prefetch next tile's rows behind this tile's WMMAs
    int ntile = tile + stride;
    if (ntile * 16 < E) {
      const float* ps = src_feat + (size_t)(ntile * 16 + n) * NODE_DIM;
      __builtin_prefetch(ps + half * 64, 0, 0);
      __builtin_prefetch(ps + half * 64 + 32, 0, 0);
      __builtin_prefetch(edge_feat + (size_t)(ntile * 16 + n) * EDGE_DIM, 0, 0);
    }

    // ----- build A fragments (16 edges x 192 padded msg cols) -------------
    int ae = min(tile * 16 + n, E - 1);          // A: M = lane&15
    float tdt = __builtin_nontemporal_load(td + ae);
    v16bf a[NKC_MSG];
#pragma unroll
    for (int kc = 0; kc < NKC_MSG; ++kc) {
#pragma unroll
      for (int ri = 0; ri < 2; ++ri) {
        int base = kc * 32 + ri * 16 + half * 8; // multiple of 8
        float v8[8];
        if (kc < 4) {                            // src_feat region [0,128)
          load8_nt(src_feat + (size_t)ae * NODE_DIM + base, v8);
        } else if (kc == 4) {                    // edge_feat region [128,160)
          load8_nt(edge_feat + (size_t)ae * EDGE_DIM + (base - NODE_DIM), v8);
        } else {                                 // time enc [160,176) + pad
#pragma unroll
          for (int i = 0; i < 8; ++i) {
            int c = base + i;
            float x = 0.0f;
            if (c < MSG_DIM) {
              int ti = c - (NODE_DIM + EDGE_DIM);
              x = tdt * te_w[ti] + te_phi[ti];
              if (ti > 0) x = __sinf(x);
            }
            v8[i] = x;
          }
        }
#pragma unroll
        for (int i = 0; i < 8; ++i) a[kc][ri * 8 + i] = (__bf16)v8[i];
      }
    }

    // dst ids for this lane's 8 output rows (D: M = r + 8*half)
    int dsts[8];
#pragma unroll
    for (int r = 0; r < 8; ++r)
      dsts[r] = dst_ids[min(tile * 16 + r + 8 * half, E - 1)];

    // ----- K pass fused with Q.K scoring -----------------------------------
    float sc[4][8];
#pragma unroll
    for (int h = 0; h < 4; ++h)
#pragma unroll
      for (int r = 0; r < 8; ++r) sc[h][r] = 0.0f;

#pragma unroll
    for (int j = 0; j < JT; ++j) {
      v8f acc = {};
#pragma unroll
      for (int kc = 0; kc < NKC_MSG; ++kc)
        acc = __builtin_amdgcn_wmma_f32_16x16x32_bf16(
            false, a[kc], false, sWk[(j * NKC_MSG + kc) * 32 + lane],
            (short)0, acc, false, false);
      float kb = Wk_b[j * 16 + n];
      int h = j >> 1;                            // head = channel/32
#pragma unroll
      for (int r = 0; r < 8; ++r) {
        float q = Qt[(size_t)dsts[r] * OUT_DIM + j * 16 + n];
        sc[h][r] += (acc[r] + kb) * q;
      }
    }

    // reduce over the 16 lanes of each half, clip/exp/mean-heads
    float am[8];
#pragma unroll
    for (int r = 0; r < 8; ++r) {
      float s = 0.0f;
#pragma unroll
      for (int h = 0; h < 4; ++h) {
        float v = sc[h][r];
        v += __shfl_xor(v, 1);
        v += __shfl_xor(v, 2);
        v += __shfl_xor(v, 4);
        v += __shfl_xor(v, 8);                   // stays within 16-lane half
        v *= inv_sqrt_hd;
        v = fminf(5.0f, fmaxf(-5.0f, v));
        s += __expf(v);
      }
      int e = tile * 16 + r + 8 * half;
      am[r] = (e < E) ? 0.25f * s : 0.0f;
    }
    if (n == 0) {                                // lane0 -> m0..7, lane16 -> m8..15
#pragma unroll
      for (int r = 0; r < 8; ++r) atomicAdd(&attn_norm[dsts[r]], am[r]);
    }

    // ----- V pass + weighted segment-sum -----------------------------------
#pragma unroll
    for (int j = 0; j < JT; ++j) {
      v8f acc = {};
#pragma unroll
      for (int kc = 0; kc < NKC_MSG; ++kc)
        acc = __builtin_amdgcn_wmma_f32_16x16x32_bf16(
            false, a[kc], false, sWv[(j * NKC_MSG + kc) * 32 + lane],
            (short)0, acc, false, false);
      float vb = Wv_b[j * 16 + n];
#pragma unroll
      for (int r = 0; r < 8; ++r)
        atomicAdd(&dst_agg[(size_t)dsts[r] * OUT_DIM + j * 16 + n],
                  (acc[r] + vb) * am[r]);
    }
  }
}

// ---------------------------------------------------------------------------
// Kernel 3: normalize by attention mass + LayerNorm. One wave32 per row.
// ---------------------------------------------------------------------------
__global__ __launch_bounds__(256) void ln_kernel(
    const float* __restrict__ agg, const float* __restrict__ nrm,
    const float* __restrict__ g, const float* __restrict__ b,
    float* __restrict__ out, int n_dst) {
  int lane = threadIdx.x & 31, wv = threadIdx.x >> 5;
  int row = blockIdx.x * 8 + wv;
  if (row >= n_dst) return;
  float inv = 1.0f / (nrm[row] + 1e-8f);
  float x[4], s = 0.0f, s2 = 0.0f;
#pragma unroll
  for (int k = 0; k < 4; ++k) {
    x[k] = __builtin_nontemporal_load(agg + (size_t)row * OUT_DIM + k * 32 + lane) * inv;
    s += x[k];
    s2 += x[k] * x[k];
  }
#pragma unroll
  for (int m = 16; m >= 1; m >>= 1) {
    s += __shfl_xor(s, m);
    s2 += __shfl_xor(s2, m);
  }
  float mean = s * (1.0f / OUT_DIM);
  float var = s2 * (1.0f / OUT_DIM) - mean * mean;
  float rstd = rsqrtf(var + 1e-5f);
#pragma unroll
  for (int k = 0; k < 4; ++k) {
    int ch = k * 32 + lane;
    __builtin_nontemporal_store((x[k] - mean) * rstd * g[ch] + b[ch],
                                out + (size_t)row * OUT_DIM + ch);
  }
}

// ---------------------------------------------------------------------------
extern "C" void kernel_launch(void* const* d_in, const int* in_sizes, int n_in,
                              void* d_out, int out_size, void* d_ws, size_t ws_size,
                              hipStream_t stream) {
  (void)n_in; (void)out_size;
  const float* src_feat  = (const float*)d_in[0];
  const float* dst_feat  = (const float*)d_in[1];
  const float* edge_feat = (const float*)d_in[2];
  const float* td        = (const float*)d_in[3];
  /* d_in[4] query_time is unused by the reference */
  const float* te_w      = (const float*)d_in[5];
  const float* te_phi    = (const float*)d_in[6];
  const float* Wq_w      = (const float*)d_in[7];
  const float* Wq_b      = (const float*)d_in[8];
  const float* Wk_w      = (const float*)d_in[9];
  const float* Wk_b      = (const float*)d_in[10];
  const float* Wv_w      = (const float*)d_in[11];
  const float* Wv_b      = (const float*)d_in[12];
  const float* ln_g      = (const float*)d_in[13];
  const float* ln_b      = (const float*)d_in[14];
  const int*   dst_ids   = (const int*)d_in[15];
  const int E     = in_sizes[3];
  const int n_dst = in_sizes[4];

  char* ws = (char*)d_ws;
  size_t off = 0;
  auto alloc = [&](size_t bytes) -> char* {
    char* p = ws + off;
    off = (off + bytes + 255) & ~(size_t)255;
    return p;
  };
  v16bf* wkf = (v16bf*)alloc((size_t)JT * NKC_MSG * 32 * 32);
  v16bf* wvf = (v16bf*)alloc((size_t)JT * NKC_MSG * 32 * 32);
  v16bf* wqf = (v16bf*)alloc((size_t)JT * NKC_Q * 32 * 32);
  float* bq  = (float*)alloc(OUT_DIM * 4);
  float* Qt  = (float*)alloc((size_t)n_dst * OUT_DIM * 4);
  float* agg = (float*)alloc((size_t)n_dst * OUT_DIM * 4);
  float* nrm = (float*)alloc((size_t)n_dst * 4);
  if (off > ws_size) return;   // workspace too small: bail (no-op)

  (void)hipMemsetAsync(agg, 0, (size_t)n_dst * OUT_DIM * 4, stream);
  (void)hipMemsetAsync(nrm, 0, (size_t)n_dst * 4, stream);

  const int prepN = 2 * (JT * NKC_MSG * 32) + JT * NKC_Q * 32 + OUT_DIM;
  prep_kernel<<<(prepN + 255) / 256, 256, 0, stream>>>(
      Wq_w, Wq_b, Wk_w, Wv_w, te_phi, wkf, wvf, wqf, bq);

  const int qtiles = (n_dst + 15) / 16;
  qtable_kernel<<<(qtiles + 7) / 8, 256, 0, stream>>>(dst_feat, wqf, bq, Qt, n_dst);

  const int etiles = (E + 15) / 16;
  int eblocks = (etiles + 7) / 8;
  if (eblocks > 1024) eblocks = 1024;            // persistent: amortize LDS fill
  edge_kernel<<<eblocks, 256, 0, stream>>>(
      src_feat, edge_feat, td, te_w, te_phi, wkf, wvf, Wk_b, Wv_b,
      dst_ids, Qt, agg, nrm, E);

  ln_kernel<<<(n_dst + 7) / 8, 256, 0, stream>>>(agg, nrm, ln_g, ln_b,
                                                 (float*)d_out, n_dst);
}
